// YOLOLoss_79834852098326
// MI455X (gfx1250) — compile-verified, matrix operand order checked by reading, not run on hardware
//
#include <hip/hip_runtime.h>
#include <math.h>

// ---------------------------------------------------------------------------
// YOLO loss for gfx1250 (MI455X).
//   K1 zero_ws      : zero mask planes + float accumulators in d_ws
//   K2 scatter      : build tobj mask (byte per cell) + n_valid per scale
//   K3 obj_wmma x3  : one block per objectness channel-plane; b128 loads,
//                     bce(x,mask), reduce via 2x v_wmma_f32_16x16x32_f16
//                     (all-ones B matrix) into f32 accumulators
//   K4 row_kernel   : per (target,anchor,scale) row: box IoU + focal cls loss
//   K5 finalize     : combine accumulators -> out[84]
// ---------------------------------------------------------------------------

#define NC   80
#define NA   3
#define NCH  255            // NA*(NC+5)
#define EPSF 1e-07f

typedef __attribute__((ext_vector_type(16))) _Float16 v16h;
typedef __attribute__((ext_vector_type(8)))  float    v8f;

// accumulator layout (floats) at d_ws offset 0
#define ACC_OBJ   0          // [3] raw wmma sums (16x inflated)
#define ACC_CLS   3          // [3] sum of row_sum
#define ACC_BOX   6          // [3] sum of (1-iou)
#define ACC_NV    9          // [3] n_valid (float)
#define ACC_SEGS  12         // [3][80] seg_sum
#define ACC_SEGC  252        // [3][80] seg_cnt
#define MASK_OFF  2048       // byte offset of tobj masks inside d_ws

__device__ const float ANCH[3][3][2] = {
    {{10.f, 13.f}, {16.f, 30.f}, {33.f, 23.f}},
    {{30.f, 61.f}, {62.f, 45.f}, {59.f, 119.f}},
    {{116.f, 90.f}, {156.f, 198.f}, {373.f, 326.f}}};

__device__ __forceinline__ float bce_logits(float x, float t) {
    return fmaxf(x, 0.0f) - x * t + log1pf(expf(-fabsf(x)));
}
__device__ __forceinline__ float sigmoidf(float x) {
    return 1.0f / (1.0f + expf(-x));
}

// ---------------------------------------------------------------------------
__global__ void zero_ws(unsigned int* ws, int nwords) {
    for (int i = blockIdx.x * blockDim.x + threadIdx.x; i < nwords;
         i += gridDim.x * blockDim.x)
        ws[i] = 0u;
}

// ---------------------------------------------------------------------------
// One thread per (scale, target): mark tobj cells, count n_valid.
__global__ void scatter_kernel(const float* __restrict__ tg, int N, int B,
                               unsigned char* __restrict__ masks,
                               float* __restrict__ acc) {
    int t = blockIdx.x * blockDim.x + threadIdx.x;
    if (t >= 3 * N) return;
    int s = t / N;
    int n = t - s * N;
    const int H[3] = {80, 40, 20};
    int h = H[s], w = H[s];
    size_t moff = 0;
    if (s >= 1) moff += (size_t)B * NA * 80 * 80;
    if (s >= 2) moff += (size_t)B * NA * 40 * 40;

    const float* T = tg + (size_t)n * 6;
    float gx = T[2] * (float)w;
    float gy = T[3] * (float)h;
    bool valid = (gx >= 0.0f) && (gx < (float)w) && (gy >= 0.0f) && (gy < (float)h);
    if (!valid) return;

    int gi = min(max((int)floorf(gx), 0), w - 1);
    int gj = min(max((int)floorf(gy), 0), h - 1);
    int b  = min(max((int)T[0], 0), B - 1);

    unsigned char* m = masks + moff;
#pragma unroll
    for (int a = 0; a < NA; ++a)
        m[(((size_t)b * NA + a) * h + gj) * w + gi] = 1;
    atomicAdd(&acc[ACC_NV + s], 3.0f);  // NA valid rows per valid target
}

// ---------------------------------------------------------------------------
// Load a 16-element chunk of one channel-plane (chunks are 16-aligned and
// 16 | hw, so a chunk never crosses the plane end when start < hw),
// compute bce(x, mask), pack to v16h. Out-of-range chunks pack zeros.
__device__ __forceinline__ v16h load_bce16(const float* __restrict__ src,
                                           const unsigned char* __restrict__ msk,
                                           int start, int hw) {
    v16h a;
#pragma unroll
    for (int i = 0; i < 16; ++i) a[i] = (_Float16)0.0f;
    if (start < hw) {
        const float4* s4 = (const float4*)(src + start);
        uint4 mw4 = *(const uint4*)(msk + start);
        unsigned mw[4] = {mw4.x, mw4.y, mw4.z, mw4.w};
        float xv[16];
        *(float4*)&xv[0]  = s4[0];
        *(float4*)&xv[4]  = s4[1];
        *(float4*)&xv[8]  = s4[2];
        *(float4*)&xv[12] = s4[3];
#pragma unroll
        for (int i = 0; i < 16; ++i) {
            float t = (float)((mw[i >> 2] >> ((i & 3) * 8)) & 1u);
            a[i] = (_Float16)bce_logits(xv[i], t);
        }
    }
    return a;
}

// Objectness bce reduction on the matrix pipe.
// blockIdx.x = channel plane (b*NA + anchor); 8 waves x 1024 elems covers
// hw <= 8192 for every scale. WMMAs run with full EXEC (uniform control flow).
__global__ void obj_wmma(const float* __restrict__ pred,
                         const unsigned char* __restrict__ mask,
                         float* __restrict__ acc, int scale, int hw) {
    const int plane = blockIdx.x;          // uniform
    const int b  = plane / NA;             // scalar division, once
    const int an = plane - b * NA;
    const float* src = pred + ((size_t)b * NCH + (size_t)an * 85 + 4) * hw;
    const unsigned char* msk = mask + (size_t)plane * hw;

    v16h ones;
#pragma unroll
    for (int i = 0; i < 16; ++i) ones[i] = (_Float16)1.0f;

    const int lane = threadIdx.x & 31;
    const int wv   = threadIdx.x >> 5;
    const int s0   = wv * 1024 + lane * 16;
    const int s1   = s0 + 512;

    // speculative prefetch of the next anchor plane's slice (safe if OOB)
    __builtin_prefetch(src + (size_t)85 * hw + s0, 0, 1);

    v16h a0 = load_bce16(src, msk, s0, hw);
    v16h a1 = load_bce16(src, msk, s1, hw);

    v8f c0 = {};
    v8f c1 = {};
    c0 = __builtin_amdgcn_wmma_f32_16x16x32_f16(
             false, a0, false, ones, (short)0, c0, false, false);
    c1 = __builtin_amdgcn_wmma_f32_16x16x32_f16(
             false, a1, false, ones, (short)0, c1, false, false);

    float s = (c0[0] + c0[1] + c0[2] + c0[3] + c0[4] + c0[5] + c0[6] + c0[7]) +
              (c1[0] + c1[1] + c1[2] + c1[3] + c1[4] + c1[5] + c1[6] + c1[7]);
    __shared__ float red[256];
    red[threadIdx.x] = s;
    __syncthreads();
    for (int off = blockDim.x >> 1; off > 0; off >>= 1) {
        if ((int)threadIdx.x < off) red[threadIdx.x] += red[threadIdx.x + off];
        __syncthreads();
    }
    if (threadIdx.x == 0) atomicAdd(&acc[ACC_OBJ + scale], red[0]);
}

// ---------------------------------------------------------------------------
// One thread per (scale, target, anchor) row: box + focal class loss.
__global__ void row_kernel(const float* __restrict__ p0,
                           const float* __restrict__ p1,
                           const float* __restrict__ p2,
                           const float* __restrict__ tg, int N, int B,
                           float* __restrict__ acc) {
    int t = blockIdx.x * blockDim.x + threadIdx.x;
    if (t >= 3 * N * NA) return;
    int s = t / (N * NA);
    int r = t - s * (N * NA);
    int n = r / NA;
    int a = r - n * NA;

    const int H[3] = {80, 40, 20};
    int h = H[s], w = H[s], hw = h * w;
    const float* pred = (s == 0) ? p0 : ((s == 1) ? p1 : p2);

    const float* T = tg + (size_t)n * 6;
    float gx = T[2] * (float)w, gy = T[3] * (float)h;
    float gw = T[4] * (float)w, gh = T[5] * (float)h;
    bool valid = (gx >= 0.0f) && (gx < (float)w) && (gy >= 0.0f) && (gy < (float)h);
    if (!valid) return;

    int gi  = min(max((int)floorf(gx), 0), w - 1);
    int gj  = min(max((int)floorf(gy), 0), h - 1);
    int b   = min(max((int)T[0], 0), B - 1);
    int cls = min(max((int)T[1], 0), NC - 1);

    const float* ps = pred + ((size_t)b * NCH + (size_t)a * 85) * hw
                           + (size_t)gj * w + gi;

    // ------ box loss: plain IoU on (cx,cy,w,h) ------
    float x0 = ps[0], x1 = ps[(size_t)hw], x2 = ps[(size_t)2 * hw], x3 = ps[(size_t)3 * hw];
    float pcx = sigmoidf(x0) * 2.0f - 0.5f;
    float pcy = sigmoidf(x1) * 2.0f - 0.5f;
    float sw  = sigmoidf(x2) * 2.0f;
    float sh  = sigmoidf(x3) * 2.0f;
    float pw  = sw * sw * ANCH[s][a][0];
    float ph  = sh * sh * ANCH[s][a][1];
    float tcx = gx - floorf(gx), tcy = gy - floorf(gy);

    float b1x1 = pcx - pw * 0.5f, b1x2 = pcx + pw * 0.5f;
    float b1y1 = pcy - ph * 0.5f, b1y2 = pcy + ph * 0.5f;
    float b2x1 = tcx - gw * 0.5f, b2x2 = tcx + gw * 0.5f;
    float b2y1 = tcy - gh * 0.5f, b2y2 = tcy + gh * 0.5f;
    float iw = fmaxf(fminf(b1x2, b2x2) - fmaxf(b1x1, b2x1), 0.0f);
    float ih = fmaxf(fminf(b1y2, b2y2) - fmaxf(b1y1, b2y1), 0.0f);
    float inter = iw * ih;
    float w1 = b1x2 - b1x1, h1 = (b1y2 - b1y1) + EPSF;
    float w2 = b2x2 - b2x1, h2 = (b2y2 - b2y1) + EPSF;
    float uni = w1 * h1 + w2 * h2 - inter + EPSF;
    float iou = inter / uni;
    atomicAdd(&acc[ACC_BOX + s], 1.0f - iou);

    // ------ focal class loss over 80 logits ------
    // focal = (1 - e^{-ce})^1.5 * ce, and om = 1-e^{-ce} in [0,1],
    // so om^1.5 = om * sqrt(om): one v_sqrt_f32 instead of a powf chain.
    float row_sum = 0.0f;
    const float* pcur = ps + (size_t)5 * hw;
#pragma unroll 4
    for (int cch = 0; cch < NC; ++cch) {
        float x  = *pcur;
        pcur += hw;
        float tt = (cch == cls) ? 1.0f : 0.0f;
        float ce = bce_logits(x, tt);
        float om = 1.0f - expf(-ce);
        row_sum += om * sqrtf(om) * ce;
    }
    atomicAdd(&acc[ACC_CLS + s], row_sum);
    atomicAdd(&acc[ACC_SEGS + s * NC + cls], row_sum);
    atomicAdd(&acc[ACC_SEGC + s * NC + cls], 1.0f);
}

// ---------------------------------------------------------------------------
__global__ void finalize_kernel(const float* __restrict__ acc,
                                float* __restrict__ out, int B) {
    const int HW[3] = {6400, 1600, 400};
    int tid = threadIdx.x;
    if (tid == 0) {
        float lb = 0.0f, lc = 0.0f, lo = 0.0f;
        for (int s = 0; s < 3; ++s) {
            float cnt = (float)(B * NA * HW[s]);
            lo += acc[ACC_OBJ + s] / (16.0f * cnt) * 1.0f;   // OBJ_GAIN; /16 wmma replication
            float nv = acc[ACC_NV + s];
            if (nv > 0.0f) {
                lb += acc[ACC_BOX + s] / fmaxf(nv, 1.0f) * 7.5f;               // BOX_GAIN
                lc += acc[ACC_CLS + s] / (fmaxf(nv, 1.0f) * (float)NC) * 0.5f; // CLS_GAIN
            }
        }
        out[0] = lb + lc + lo;
        out[1] = lb;
        out[2] = lc;
        out[3] = lo;
    }
    if (tid < NC) {
        float pc = 0.0f;
        for (int s = 0; s < 3; ++s) {
            float cnt = acc[ACC_SEGC + s * NC + tid];
            if (cnt > 0.0f)
                pc += acc[ACC_SEGS + s * NC + tid] / (cnt * (float)NC) * 0.5f;
        }
        out[4 + tid] = pc;
    }
}

// ---------------------------------------------------------------------------
extern "C" void kernel_launch(void* const* d_in, const int* in_sizes, int n_in,
                              void* d_out, int out_size, void* d_ws, size_t ws_size,
                              hipStream_t stream) {
    const float* p0 = (const float*)d_in[0];
    const float* p1 = (const float*)d_in[1];
    const float* p2 = (const float*)d_in[2];
    const float* tg = (const float*)d_in[3];

    int B = in_sizes[0] / (NCH * 80 * 80);   // 16
    int N = in_sizes[3] / 6;                  // 800

    float* acc = (float*)d_ws;
    unsigned char* masks = (unsigned char*)d_ws + MASK_OFF;

    int maskBytes = B * NA * (6400 + 1600 + 400);
    int zeroWords = (MASK_OFF + maskBytes + 3) / 4;
    zero_ws<<<(zeroWords + 255) / 256, 256, 0, stream>>>((unsigned int*)d_ws,
                                                         zeroWords);

    scatter_kernel<<<(3 * N + 255) / 256, 256, 0, stream>>>(tg, N, B, masks, acc);

    const float* preds[3] = {p0, p1, p2};
    const int hws[3] = {6400, 1600, 400};
    size_t moff = 0;
    for (int s = 0; s < 3; ++s) {
        // one block per objectness channel-plane; 8 waves x 1024 >= hw
        obj_wmma<<<B * NA, 256, 0, stream>>>(preds[s], masks + moff, acc, s,
                                             hws[s]);
        moff += (size_t)B * NA * hws[s];
    }

    int rows = 3 * N * NA;
    row_kernel<<<(rows + 255) / 256, 256, 0, stream>>>(p0, p1, p2, tg, N, B, acc);

    finalize_kernel<<<1, 128, 0, stream>>>(acc, (float*)d_out, B);
}
